// MaskedMHA_27169963115063
// MI455X (gfx1250) — matrix operand order repaired and checked by reading, not compiled, not run on hardware
//
#include <hip/hip_runtime.h>
#include <stddef.h>
#include <stdint.h>

// ---------------------------------------------------------------------------
// Masked MHA for MI455X (gfx1250, wave32, WMMA bf16 16x16x32, f32 accum).
// Pipeline: cvt weights -> transpose x -> fused QKV GEMM (async-to-LDS weight
// staging, double buffered) -> flash attention (async-to-LDS K/V staging,
// double buffered) -> output projection (register-pipelined).
// Workgroup syncs use raw split barriers (s_barrier_signal/-wait) so vector
// loads stay in flight across iterations (no full waitcnt drain).
// ---------------------------------------------------------------------------

typedef __bf16 v16bf __attribute__((ext_vector_type(16)));
typedef __bf16 v8bf  __attribute__((ext_vector_type(8)));
typedef float  v8f   __attribute__((ext_vector_type(8)));

#define DEV __device__ __forceinline__

namespace {
constexpr int kB = 2;
constexpr int kC = 1024;
constexpr int kT = 2048;
constexpr int kH = 16;
constexpr int kD = 64;   // kC / kH
}

DEV v8f wmma_bf16(v16bf a, v16bf b, v8f c) {
  // 8 args: (neg_a, A, neg_b, B, c_mod, C, reuse_a, reuse_b)
  return __builtin_amdgcn_wmma_f32_16x16x32_bf16(
      false, a, false, b, (short)0, c, false, false);
}

// A-matrix fragment (16x32 bf16, MxK), same K order as B fragments:
//   lanes 0-15 : row M=lane,    K 0..15
//   lanes16-31 : row M=lane-16, K 16..31
DEV v16bf load_a_frag(const __bf16* row, int hi) {
  v16bf a;
  ((v8bf*)&a)[0] = *(const v8bf*)(row + hi * 16);
  ((v8bf*)&a)[1] = *(const v8bf*)(row + hi * 16 + 8);
  return a;
}

// B-matrix fragment (32x16 bf16, KxN): lane holds column N=lane&15 with a
// contiguous run of 16 K values (lanes 0-15: K 0..15, lanes 16-31: K 16..31).
DEV v16bf load_b_frag(const __bf16* col, int hi) {
  return *(const v16bf*)(col + hi * 16);
}

// CDNA5 async copy: 16B global -> LDS, tracked by ASYNCcnt.
DEV void async_g2lds_b128(unsigned lds_off, const __bf16* gsrc) {
  asm volatile("global_load_async_to_lds_b128 %0, %1, off"
               :: "v"(lds_off), "v"(gsrc) : "memory");
}
DEV void wait_asynccnt0() {
  asm volatile("s_wait_asynccnt 0" ::: "memory");
}
DEV void wait_dscnt0() {
  asm volatile("s_wait_dscnt 0" ::: "memory");
}
// Workgroup split-barrier without the implicit full-waitcnt fence of
// __syncthreads(). Safe here: every wave drains its own ASYNCcnt before
// signaling, and all cross-wave LDS reads were already consumed by WMMAs.
DEV void block_sync_nofence() {
  asm volatile("s_barrier_signal -1\n\ts_barrier_wait -1" ::: "memory");
}

// ---------------------------------------------------------------------------
// Kernel 1: fp32 -> bf16 weight conversion
// ---------------------------------------------------------------------------
__global__ void mha_cvt_w_kernel(const float* __restrict__ w,
                                 __bf16* __restrict__ wb, int n) {
  int i = blockIdx.x * blockDim.x + threadIdx.x;
  if (i < n) wb[i] = (__bf16)w[i];
}

// ---------------------------------------------------------------------------
// Kernel 2: x (B,C,T) fp32 -> xT (B,T,C) bf16  (LDS tiled transpose)
// ---------------------------------------------------------------------------
__global__ void mha_transpose_x_kernel(const float* __restrict__ x,
                                       __bf16* __restrict__ xT) {
  __shared__ float tile[32][33];
  const int b = blockIdx.z;
  const int t0 = blockIdx.x * 32;
  const int c0 = blockIdx.y * 32;
  const int tx = threadIdx.x;       // 0..31
  const int ty = threadIdx.y;       // 0..7
#pragma unroll
  for (int i = 0; i < 32; i += 8)
    tile[ty + i][tx] =
        x[(size_t)b * kC * kT + (size_t)(c0 + ty + i) * kT + (t0 + tx)];
  __syncthreads();
#pragma unroll
  for (int i = 0; i < 32; i += 8)
    xT[(size_t)b * kT * kC + (size_t)(t0 + ty + i) * kC + (c0 + tx)] =
        (__bf16)tile[tx][ty + i];
}

// ---------------------------------------------------------------------------
// Kernel 3: fused QKV projection.
// Block: 256 thr (8 waves). o-tile = 16 rows shared by all waves; each wave
// computes a 16(o) x 32(t) tile for Q, K and V (6 WMMAs / K-step).
// Wq/Wk/Wv K-chunk tiles staged into LDS via global_load_async_to_lds_b128,
// double-buffered; x B-fragments register-pipelined across the split barrier.
// ---------------------------------------------------------------------------
__global__ void mha_qkv_kernel(const __bf16* __restrict__ WqB,
                               const __bf16* __restrict__ WkB,
                               const __bf16* __restrict__ WvB,
                               const float* __restrict__ bq,
                               const float* __restrict__ bk,
                               const float* __restrict__ bv,
                               const __bf16* __restrict__ xT,
                               const unsigned char* __restrict__ mask,
                               __bf16* __restrict__ Qh,
                               __bf16* __restrict__ Kh,
                               __bf16* __restrict__ Vt) {
  __shared__ __align__(16) __bf16 wtile[2][3][16][32];  // 6 KB
  const int tid  = threadIdx.x;
  const int lane = tid & 31;
  const int wave = tid >> 5;
  const int l16  = lane & 15;
  const int hi   = lane >> 4;
  const int o0   = blockIdx.x * 16;
  const int t0w  = blockIdx.y * 256 + wave * 32;
  const int b    = blockIdx.z;

  // --- async staging: 192 threads move 3 x 1KB per K-chunk ---
  const __bf16* Ws[3] = {WqB, WkB, WvB};
  const int sw   = tid >> 6;        // 0..3 (3 = idle)
  const int sj   = tid & 63;
  const int srow = sj >> 2;
  const int sch  = sj & 3;
  auto stage = [&](int buf, int c0) {
    if (sw < 3) {
      const __bf16* src = Ws[sw] + (size_t)(o0 + srow) * kC + c0 + sch * 8;
      unsigned dst = (unsigned)(size_t)&wtile[buf][sw][srow][sch * 8];
      async_g2lds_b128(dst, src);
    }
  };

  const int tA = t0w + l16;
  const int tB = t0w + 16 + l16;
  const __bf16* xrowA = xT + ((size_t)b * kT + tA) * kC;
  const __bf16* xrowB = xT + ((size_t)b * kT + tB) * kC;

  stage(0, 0);
  v16bf bxA = load_b_frag(xrowA, hi);
  v16bf bxB = load_b_frag(xrowB, hi);
  wait_asynccnt0();
  block_sync_nofence();

  v8f accq[2] = {}, acck[2] = {}, accv[2] = {};
  for (int c0 = 0; c0 < kC; c0 += 32) {
    const int cur = (c0 >> 5) & 1;
    const int c0n = (c0 + 32) & (kC - 1);   // wraps harmlessly on last iter
    stage(cur ^ 1, c0n);
    const v16bf bxAn = load_b_frag(xrowA + c0n, hi);
    const v16bf bxBn = load_b_frag(xrowB + c0n, hi);

    const v16bf aqf = load_a_frag(&wtile[cur][0][l16][0], hi);
    const v16bf akf = load_a_frag(&wtile[cur][1][l16][0], hi);
    const v16bf avf = load_a_frag(&wtile[cur][2][l16][0], hi);
    accq[0] = wmma_bf16(aqf, bxA, accq[0]);
    accq[1] = wmma_bf16(aqf, bxB, accq[1]);
    acck[0] = wmma_bf16(akf, bxA, acck[0]);
    acck[1] = wmma_bf16(akf, bxB, acck[1]);
    accv[0] = wmma_bf16(avf, bxA, accv[0]);
    accv[1] = wmma_bf16(avf, bxB, accv[1]);

    wait_asynccnt0();
    block_sync_nofence();
    bxA = bxAn;
    bxB = bxBn;
  }

  // C/D layout: element g of lane -> M = g + 8*hi (o dim), N = l16 (t dim).
  const int h  = o0 >> 6;
  const int d0 = (o0 & 63) + 8 * hi;  // 16-tile never straddles a head
  const size_t headT = ((size_t)b * kH + h) * (size_t)kT;
  const size_t headD = ((size_t)b * kH + h) * (size_t)kD;

#pragma unroll
  for (int j = 0; j < 2; ++j) {
    const int t = t0w + j * 16 + l16;
    v8bf qs, ks;
#pragma unroll
    for (int g = 0; g < 8; ++g) {
      const int o = o0 + 8 * hi + g;
      qs[g] = (__bf16)((accq[j][g] + bq[o]) * 0.125f);  // 1/sqrt(64)
      ks[g] = (__bf16)(acck[j][g] + bk[o]);
    }
    *(v8bf*)(Qh + (headT + t) * kD + d0) = qs;
    *(v8bf*)(Kh + (headT + t) * kD + d0) = ks;
    const float mval = mask[(size_t)b * kT + t] ? 1.0f : 0.0f;
#pragma unroll
    for (int g = 0; g < 8; ++g) {
      const int o = o0 + 8 * hi + g;
      Vt[(headD + (d0 + g)) * kT + t] = (__bf16)((accv[j][g] + bv[o]) * mval);
    }
  }
}

// ---------------------------------------------------------------------------
// Kernel 4: flash attention. Block = 8 waves sharing one (b,h); each wave
// owns a 16-row Q tile. Per 32-wide K-tile: the 32x64 K tile and 64x32 V tile
// are staged into LDS with async copies (double-buffered; all 256 threads
// issue 2 transfers each). Row padding (144B / 80B strides) gives conflict-
// free 16-row ds_load_b128 B-fragments. 4 WMMAs for S, online softmax with
// __shfl_xor butterflies, P through per-wave LDS, 4 WMMAs for PV.
// ---------------------------------------------------------------------------
__global__ void mha_attn_kernel(const __bf16* __restrict__ Qh,
                                const __bf16* __restrict__ Kh,
                                const __bf16* __restrict__ Vt,
                                const unsigned char* __restrict__ mask,
                                __bf16* __restrict__ Oh) {
  __shared__ __align__(16) __bf16 ktile[2][32][72];  //  9,216 B
  __shared__ __align__(16) __bf16 vtile[2][64][40];  // 10,240 B
  __shared__ __align__(16) __bf16 psmem[8][16][40];  // 10,240 B
  const int tid  = threadIdx.x;
  const int lane = tid & 31;
  const int wave = tid >> 5;
  const int l16  = lane & 15;
  const int hi   = lane >> 4;
  const int h    = blockIdx.y;
  const int b    = blockIdx.z;
  const int tq0  = blockIdx.x * 128 + wave * 16;

  const size_t headTD = ((size_t)b * kH + h) * (size_t)kT * kD;
  const __bf16* qrow  = Qh + headTD + (size_t)(tq0 + l16) * kD;
  const __bf16* Khead = Kh + headTD;
  const __bf16* Vhead = Vt + headTD;                 // (D,T) layout
  const unsigned char* mrow = mask + (size_t)b * kT;

  // Staging assignments: K = 32 rows x 8 chunks, V = 64 rows x 4 chunks.
  const int krow = tid >> 3, kch = (tid & 7) * 8;
  const int vrow = tid >> 2, vch = (tid & 3) * 8;
  auto stage = [&](int buf, int k0) {
    async_g2lds_b128((unsigned)(size_t)&ktile[buf][krow][kch],
                     Khead + (size_t)(k0 + krow) * kD + kch);
    async_g2lds_b128((unsigned)(size_t)&vtile[buf][vrow][vch],
                     Vhead + (size_t)vrow * kT + k0 + vch);
  };

  const v16bf qa0 = load_a_frag(qrow + 0, hi);
  const v16bf qa1 = load_a_frag(qrow + 32, hi);

  float m[8], l[8];
  v8f acc[4] = {};
#pragma unroll
  for (int g = 0; g < 8; ++g) { m[g] = -3.0e38f; l[g] = 0.0f; }

  stage(0, 0);
  wait_asynccnt0();
  block_sync_nofence();

  for (int k0 = 0; k0 < kT; k0 += 32) {
    const int cur = (k0 >> 5) & 1;
    const int k0n = (k0 + 32) & (kT - 1);   // wraps on last iter (unused data)
    stage(cur ^ 1, k0n);

    // ---- S = (Q*scale) K^T, 16 x 32, K-fragments from LDS ----
    const __bf16* kr0 = &ktile[cur][l16][0];
    const __bf16* kr1 = &ktile[cur][16 + l16][0];
    v8f S0 = {}, S1 = {};
    S0 = wmma_bf16(qa0, load_b_frag(kr0 + 0, hi), S0);
    S0 = wmma_bf16(qa1, load_b_frag(kr0 + 32, hi), S0);
    S1 = wmma_bf16(qa0, load_b_frag(kr1 + 0, hi), S1);
    S1 = wmma_bf16(qa1, load_b_frag(kr1 + 32, hi), S1);

    // V fragments from LDS (issued before the softmax VALU block).
    v16bf vb[4];
#pragma unroll
    for (int dt = 0; dt < 4; ++dt)
      vb[dt] = load_b_frag(&vtile[cur][dt * 16 + l16][0], hi);

    const bool ok0 = mrow[k0 + l16] != 0;
    const bool ok1 = mrow[k0 + 16 + l16] != 0;

    // ---- online softmax per row (row M = g + 8*hi, cols across lanes) ----
#pragma unroll
    for (int g = 0; g < 8; ++g) {
      float s0 = ok0 ? S0[g] : -3.0e38f;
      float s1 = ok1 ? S1[g] : -3.0e38f;
      float r = fmaxf(s0, s1);
#pragma unroll
      for (int msk = 8; msk >= 1; msk >>= 1)
        r = fmaxf(r, __shfl_xor(r, msk, 32));      // stays within 16-lane half
      const float mnew = fmaxf(m[g], r);
      const float corr = __expf(m[g] - mnew);      // finite by construction
      const float p0 = ok0 ? __expf(s0 - mnew) : 0.0f;
      const float p1 = ok1 ? __expf(s1 - mnew) : 0.0f;
      float rs = p0 + p1;
#pragma unroll
      for (int msk = 8; msk >= 1; msk >>= 1)
        rs += __shfl_xor(rs, msk, 32);
      l[g] = l[g] * corr + rs;
      m[g] = mnew;
      acc[0][g] *= corr; acc[1][g] *= corr;
      acc[2][g] *= corr; acc[3][g] *= corr;
      psmem[wave][g + 8 * hi][l16]      = (__bf16)p0;
      psmem[wave][g + 8 * hi][16 + l16] = (__bf16)p1;
    }
    wait_dscnt0();  // per-wave LDS RAW on the split DS counter (CDNA5)

    // ---- PV: out(16x64) += P(16x32) * V(32x64) ----
    const v16bf pa = load_a_frag(&psmem[wave][l16][0], hi);
#pragma unroll
    for (int dt = 0; dt < 4; ++dt)
      acc[dt] = wmma_bf16(pa, vb[dt], acc[dt]);

    wait_asynccnt0();
    block_sync_nofence();
  }

  // ---- normalize and store (B,H,T,D) ----
#pragma unroll
  for (int g = 0; g < 8; ++g) {
    const float inv = (l[g] > 0.0f) ? 1.0f / l[g] : 0.0f;  // all-masked -> 0
    __bf16* orow = Oh + headTD + (size_t)(tq0 + g + 8 * hi) * kD;
#pragma unroll
    for (int dt = 0; dt < 4; ++dt)
      orow[dt * 16 + l16] = (__bf16)(acc[dt][g] * inv);
  }
}

// ---------------------------------------------------------------------------
// Kernel 5: output projection  out[b,o,t] = (Wp . Oc + bp[o]) * mask[t]
// Wave computes 16(o) x 32(t); fragments register-pipelined across the K loop.
// ---------------------------------------------------------------------------
__global__ void mha_out_proj_kernel(const __bf16* __restrict__ WpB,
                                    const float* __restrict__ bp,
                                    const __bf16* __restrict__ Oh,
                                    const unsigned char* __restrict__ mask,
                                    float* __restrict__ out) {
  const int lane = threadIdx.x & 31;
  const int wave = threadIdx.x >> 5;
  const int l16  = lane & 15;
  const int hi   = lane >> 4;
  const int o0   = blockIdx.x * 16;
  const int t0w  = blockIdx.y * 256 + wave * 32;
  const int b    = blockIdx.z;
  const int tA   = t0w + l16;
  const int tB   = t0w + 16 + l16;

  const __bf16* wr    = WpB + (size_t)(o0 + l16) * kC;
  const __bf16* obase = Oh + (size_t)b * kH * kT * kD;
  auto bcol = [&](int c, int t) {
    return obase + ((size_t)(c >> 6) * kT + t) * kD + (c & 63);
  };

  v8f acc0 = {}, acc1 = {};
  v16bf a_c  = load_a_frag(wr, hi);
  v16bf b0_c = *(const v16bf*)bcol(hi * 16, tA);
  v16bf b1_c = *(const v16bf*)bcol(hi * 16, tB);
  for (int c0 = 0; c0 < kC; c0 += 32) {
    const int c0n = (c0 + 32) & (kC - 1);
    const v16bf a_n  = load_a_frag(wr + c0n, hi);
    const v16bf b0_n = *(const v16bf*)bcol(c0n + hi * 16, tA);
    const v16bf b1_n = *(const v16bf*)bcol(c0n + hi * 16, tB);
    acc0 = wmma_bf16(a_c, b0_c, acc0);
    acc1 = wmma_bf16(a_c, b1_c, acc1);
    a_c = a_n; b0_c = b0_n; b1_c = b1_n;
  }

  const float mA = mask[(size_t)b * kT + tA] ? 1.0f : 0.0f;
  const float mB = mask[(size_t)b * kT + tB] ? 1.0f : 0.0f;
#pragma unroll
  for (int g = 0; g < 8; ++g) {
    const int o = o0 + 8 * hi + g;
    out[((size_t)b * kC + o) * kT + tA] = (acc0[g] + bp[o]) * mA;
    out[((size_t)b * kC + o) * kT + tB] = (acc1[g] + bp[o]) * mB;
  }
}

// ---------------------------------------------------------------------------
// Kernel 6: second tuple output — mask as float, appended after out.
// ---------------------------------------------------------------------------
__global__ void mha_mask_tail_kernel(const unsigned char* __restrict__ mask,
                                     float* __restrict__ tail, int n) {
  int i = blockIdx.x * blockDim.x + threadIdx.x;
  if (i < n) tail[i] = mask[i] ? 1.0f : 0.0f;
}

// ---------------------------------------------------------------------------
extern "C" void kernel_launch(void* const* d_in, const int* in_sizes, int n_in,
                              void* d_out, int out_size, void* d_ws,
                              size_t ws_size, hipStream_t stream) {
  (void)in_sizes; (void)n_in; (void)out_size; (void)ws_size;
  const float* x  = (const float*)d_in[0];
  const unsigned char* mask = (const unsigned char*)d_in[1];  // jax bool = 1B
  const float* Wq = (const float*)d_in[2];
  const float* bq = (const float*)d_in[3];
  const float* Wk = (const float*)d_in[4];
  const float* bk = (const float*)d_in[5];
  const float* Wv = (const float*)d_in[6];
  const float* bv = (const float*)d_in[7];
  const float* Wp = (const float*)d_in[8];
  const float* bp = (const float*)d_in[9];
  float* out = (float*)d_out;

  // Workspace layout (bf16 elements): 4 weights (1M each) + xT (4M) +
  // Qh/Kh/Vt/Oh (4M each) = 24M elems = 48 MB.
  __bf16* wsb = (__bf16*)d_ws;
  const size_t WW = (size_t)kC * kC;
  const size_t XN = (size_t)kB * kT * kC;
  const size_t QN = (size_t)kB * kH * kT * kD;
  __bf16* WqB = wsb;
  __bf16* WkB = WqB + WW;
  __bf16* WvB = WkB + WW;
  __bf16* WpB = WvB + WW;
  __bf16* xT  = WpB + WW;
  __bf16* Qh  = xT + XN;
  __bf16* Kh  = Qh + QN;
  __bf16* Vt  = Kh + QN;
  __bf16* Oh  = Vt + QN;

  const int nW = (int)WW;
  mha_cvt_w_kernel<<<dim3((nW + 255) / 256), 256, 0, stream>>>(Wq, WqB, nW);
  mha_cvt_w_kernel<<<dim3((nW + 255) / 256), 256, 0, stream>>>(Wk, WkB, nW);
  mha_cvt_w_kernel<<<dim3((nW + 255) / 256), 256, 0, stream>>>(Wv, WvB, nW);
  mha_cvt_w_kernel<<<dim3((nW + 255) / 256), 256, 0, stream>>>(Wp, WpB, nW);

  mha_transpose_x_kernel<<<dim3(kT / 32, kC / 32, kB), dim3(32, 8), 0,
                           stream>>>(x, xT);

  mha_qkv_kernel<<<dim3(kC / 16, kT / 256, kB), 256, 0, stream>>>(
      WqB, WkB, WvB, bq, bk, bv, xT, mask, Qh, Kh, Vt);

  mha_attn_kernel<<<dim3(kT / 128, kH, kB), 256, 0, stream>>>(
      Qh, Kh, Vt, mask, Oh);

  mha_out_proj_kernel<<<dim3(kC / 16, kT / 256, kB), 256, 0, stream>>>(
      WpB, bp, Oh, mask, out);

  mha_mask_tail_kernel<<<dim3((kB * kT + 255) / 256), 256, 0, stream>>>(
      mask, out + (size_t)kB * kC * kT, kB * kT);
}
